// Transformer_42898133353384
// MI455X (gfx1250) — matrix operand order repaired and checked by reading, not compiled, not run on hardware
//
#include <hip/hip_runtime.h>

typedef __bf16 bf16;
typedef __attribute__((ext_vector_type(8)))  __bf16 v8bf;
typedef __attribute__((ext_vector_type(16))) __bf16 v16bf;
typedef __attribute__((ext_vector_type(8)))  float  v8f;

#define S_LEN 2048
#define D_MODEL 1024
#define N_HEAD 16
#define D_HEAD 64
#define D_FF 512
#define N_CLS 3

// ---------------------------------------------------------------------------
// WMMA + vector-load helpers
// ---------------------------------------------------------------------------
__device__ __forceinline__ v8f wmma_bf16(v16bf a, v16bf b, v8f c) {
  return __builtin_amdgcn_wmma_f32_16x16x32_bf16(
      /*neg_a=*/false, a, /*neg_b=*/false, b,
      /*c_mod=*/(short)0, c, /*reuse_a=*/false, /*reuse_b=*/false);
}

__device__ __forceinline__ v8bf ld8(const bf16* p) { return *(const v8bf*)p; }

__device__ __forceinline__ v16bf cat8(v8bf lo, v8bf hi) {
  return __builtin_shufflevector(lo, hi, 0, 1, 2, 3, 4, 5, 6, 7,
                                 8, 9, 10, 11, 12, 13, 14, 15);
}

// 16 contiguous bf16 (two b128 loads)
__device__ __forceinline__ v16bf ld16(const bf16* p) {
  return cat8(ld8(p), ld8(p + 8));
}

// A-operand 16x32 bf16 tile. `row` = base + (lane&15)*ld.
// Per ISA 16-bit A table the lane's 16 elements are two contiguous 8-element
// runs: K = k0+grp*8 .. +7  and  K = k0+16+grp*8 .. +7  -> two b128 loads.
__device__ __forceinline__ v16bf load_a_tile(const bf16* row, int k0, int grp) {
  const bf16* p = row + k0 + grp * 8;
  return cat8(ld8(p), ld8(p + 16));
}

__device__ __forceinline__ v16bf ld_b_tile(const bf16* col, size_t ld, int grp) {
  v16bf b;
#pragma unroll
  for (int s = 0; s < 8; ++s) {
    b[s] = col[(size_t)(grp * 8 + s) * ld];
    b[8 + s] = col[(size_t)(16 + grp * 8 + s) * ld];
  }
  return b;
}

__device__ __forceinline__ float rmax16(float v) {
#pragma unroll
  for (int m = 1; m < 16; m <<= 1) v = fmaxf(v, __shfl_xor(v, m, 16));
  return v;
}
__device__ __forceinline__ float rsum16(float v) {
#pragma unroll
  for (int m = 1; m < 16; m <<= 1) v += __shfl_xor(v, m, 16);
  return v;
}

// ---------------------------------------------------------------------------
// Weight prep
// ---------------------------------------------------------------------------
__global__ void cvt_plain_kernel(const float* __restrict__ in,
                                 bf16* __restrict__ out, int n) {
  int i = blockIdx.x * blockDim.x + threadIdx.x;
  if (i < n) out[i] = (bf16)in[i];
}

// in[h][d][k] (f32) -> out[d][h*K + k] (bf16)
__global__ void cvt_qkv_kernel(const float* __restrict__ in,
                               bf16* __restrict__ out, int H, int D, int K) {
  int i = blockIdx.x * blockDim.x + threadIdx.x;
  int total = H * D * K;
  if (i >= total) return;
  int k = i % K;
  int d = (i / K) % D;
  int h = i / (K * D);
  out[(size_t)d * (H * K) + h * K + k] = (bf16)in[i];
}

// ---------------------------------------------------------------------------
// Embedding gather
// ---------------------------------------------------------------------------
__global__ void embed_kernel(const int* __restrict__ idx,
                             const float* __restrict__ emb,
                             float* __restrict__ Xf, bf16* __restrict__ Xh) {
  int s = blockIdx.x;
  int id = idx[s];
  for (int d = threadIdx.x; d < D_MODEL; d += blockDim.x) {
    float v = emb[(size_t)id * D_MODEL + d];
    Xf[(size_t)s * D_MODEL + d] = v;
    Xh[(size_t)s * D_MODEL + d] = (bf16)v;
  }
}

// ---------------------------------------------------------------------------
// WMMA GEMM: 32x64 output tile per wave (2 A tiles reuse 4 B tiles).
// grid = (M/32, N/64), block = 32.
// Optional transposed bf16 store: out_t[col*M + row] (for K^T).
// ---------------------------------------------------------------------------
template <bool BIAS, bool RELU>
__global__ void gemm_bf16_kernel(const bf16* __restrict__ A,
                                 const bf16* __restrict__ B,
                                 const float* __restrict__ bias,
                                 const float* __restrict__ residual,
                                 float* __restrict__ out_f32,
                                 bf16* __restrict__ out_bf16,
                                 bf16* __restrict__ out_bf16_t,
                                 int M, int N, int K) {
  const int lane = threadIdx.x;
  const int grp  = lane >> 4;
  const int mrow = lane & 15;
  const int mblk = blockIdx.x;
  const int nblk = blockIdx.y;

  v8f acc[2][4] = {};
  const bf16* Arow0 = A + ((size_t)(mblk * 32) + mrow) * K;
  const bf16* Arow1 = Arow0 + (size_t)16 * K;

  for (int k0 = 0; k0 < K; k0 += 32) {
    v16bf a0 = load_a_tile(Arow0, k0, grp);
    v16bf a1 = load_a_tile(Arow1, k0, grp);
    const bf16* Brow = B + (size_t)k0 * N + nblk * 64 + mrow;
#pragma unroll
    for (int t = 0; t < 4; ++t) {
      v16bf b = ld_b_tile(Brow + t * 16, N, grp);
      acc[0][t] = wmma_bf16(a0, b, acc[0][t]);
      acc[1][t] = wmma_bf16(a1, b, acc[1][t]);
    }
  }

#pragma unroll
  for (int mi = 0; mi < 2; ++mi) {
#pragma unroll
    for (int t = 0; t < 4; ++t) {
      int col = nblk * 64 + t * 16 + (lane & 15);
      float bv = BIAS ? bias[col] : 0.0f;
#pragma unroll
      for (int r = 0; r < 8; ++r) {
        int row = mblk * 32 + mi * 16 + grp * 8 + r;
        float v = acc[mi][t][r] + bv;
        if (RELU) v = v > 0.0f ? v : 0.0f;
        size_t o = (size_t)row * N + col;
        if (residual) v += residual[o];
        if (out_f32)    out_f32[o] = v;
        if (out_bf16)   out_bf16[o] = (bf16)v;
        if (out_bf16_t) out_bf16_t[(size_t)col * M + row] = (bf16)v;
      }
    }
  }
}

// ---------------------------------------------------------------------------
// Flash attention with ALiBi + causal mask.
// grid = (S/16, H), block = 32 (one wave per 16-query block per head)
// Q,V bf16 [S, H*64]; KT bf16 [H*64, S]; output Zc bf16 [S, H*64]
// ---------------------------------------------------------------------------
__global__ void flash_attn_kernel(const bf16* __restrict__ Qh,
                                  const bf16* __restrict__ KT,
                                  const bf16* __restrict__ Vh,
                                  bf16* __restrict__ Zc) {
  const int lane = threadIdx.x;
  const int grp  = lane >> 4;
  const int mrow = lane & 15;
  const int qb   = blockIdx.x;
  const int h    = blockIdx.y;
  const int Dm   = N_HEAD * D_HEAD;  // 1024
  const float invsqrt = 0.125f;      // 1/sqrt(64)
  const float slope = exp2f((-8.0f / N_HEAD) * (float)(h + 1));
  const float NEG_INF = -__builtin_inff();

  __shared__ bf16 Pld[16 * 32];

  // Q tile (16 rows x 64 head-dims) as two A operands
  const bf16* Qrow = Qh + ((size_t)(qb * 16) + mrow) * Dm + h * D_HEAD;
  v16bf aq0 = load_a_tile(Qrow, 0, grp);
  v16bf aq1 = load_a_tile(Qrow, 32, grp);

  // K^T column (key = lane&15) for this lane; contraction indices strided by S
  const bf16* KTrow0 = KT + (size_t)(h * D_HEAD) * S_LEN + mrow;
  const bf16* KTrow1 = KT + (size_t)(h * D_HEAD + 32) * S_LEN + mrow;

  v8f o[4] = {};
  float mrw[8], lrw[8], alpha[8];
#pragma unroll
  for (int r = 0; r < 8; ++r) { mrw[r] = NEG_INF; lrw[r] = 0.0f; }

  const int imax = qb * 16 + 15;
  for (int j0 = 0; j0 <= imax; j0 += 32) {
    // ---- scores: two 16x16 f32 tiles (keys j0..+15 and j0+16..+31) ----
    v8f s0 = {}, s1 = {};
    s0 = wmma_bf16(aq0, ld_b_tile(KTrow0 + j0, S_LEN, grp), s0);
    s0 = wmma_bf16(aq1, ld_b_tile(KTrow1 + j0, S_LEN, grp), s0);
    s1 = wmma_bf16(aq0, ld_b_tile(KTrow0 + j0 + 16, S_LEN, grp), s1);
    s1 = wmma_bf16(aq1, ld_b_tile(KTrow1 + j0 + 16, S_LEN, grp), s1);

    // ---- online softmax (rows spread across the 16-lane group) ----
    const int n = lane & 15;
    const int ja = j0 + n;
    const int jb = j0 + 16 + n;
#pragma unroll
    for (int r = 0; r < 8; ++r) {
      const int i = qb * 16 + grp * 8 + r;
      float f0 = (ja <= i) ? (s0[r] * invsqrt - (float)(i - ja) * slope) : NEG_INF;
      float f1 = (jb <= i) ? (s1[r] * invsqrt - (float)(i - jb) * slope) : NEG_INF;
      float mn = fmaxf(mrw[r], rmax16(fmaxf(f0, f1)));
      float a  = __expf(mrw[r] - mn);  // 0 on first block (-inf - finite)
      float p0 = __expf(f0 - mn);
      float p1 = __expf(f1 - mn);
      lrw[r] = lrw[r] * a + rsum16(p0 + p1);
      mrw[r] = mn;
      alpha[r] = a;
      const int M = grp * 8 + r;
      Pld[M * 32 + n]      = (bf16)p0;
      Pld[M * 32 + 16 + n] = (bf16)p1;
    }
#pragma unroll
    for (int t = 0; t < 4; ++t)
#pragma unroll
      for (int r = 0; r < 8; ++r) o[t][r] *= alpha[r];

    __syncthreads();

    // ---- O += P(16x32) x V(32x64) ----
    v16bf ap = load_a_tile((const bf16*)Pld + mrow * 32, 0, grp);
    const bf16* Vb = Vh + (size_t)j0 * Dm + h * D_HEAD + mrow;
#pragma unroll
    for (int t = 0; t < 4; ++t) {
      o[t] = wmma_bf16(ap, ld_b_tile(Vb + t * 16, Dm, grp), o[t]);
    }
    __syncthreads();
  }

  // ---- normalize + store ----
#pragma unroll
  for (int t = 0; t < 4; ++t) {
    int col = h * D_HEAD + t * 16 + (lane & 15);
#pragma unroll
    for (int r = 0; r < 8; ++r) {
      int row = qb * 16 + grp * 8 + r;
      Zc[(size_t)row * Dm + col] = (bf16)(o[t][r] / lrw[r]);
    }
  }
}

// ---------------------------------------------------------------------------
// Classifier: out = log_softmax(X @ Wc + bc), C=3.  One wave per row.
// ---------------------------------------------------------------------------
__global__ void classifier_kernel(const float* __restrict__ X,
                                  const float* __restrict__ Wc,
                                  const float* __restrict__ bc,
                                  float* __restrict__ out) {
  int s = blockIdx.x;
  int lane = threadIdx.x;  // 32
  float acc[N_CLS] = {0.0f, 0.0f, 0.0f};
  for (int d = lane; d < D_MODEL; d += 32) {
    float x = X[(size_t)s * D_MODEL + d];
#pragma unroll
    for (int c = 0; c < N_CLS; ++c) acc[c] += x * Wc[d * N_CLS + c];
  }
#pragma unroll
  for (int c = 0; c < N_CLS; ++c) {
#pragma unroll
    for (int m = 1; m < 32; m <<= 1) acc[c] += __shfl_xor(acc[c], m, 32);
    acc[c] += bc[c];
  }
  float mx = fmaxf(acc[0], fmaxf(acc[1], acc[2]));
  float lse = mx + logf(expf(acc[0] - mx) + expf(acc[1] - mx) + expf(acc[2] - mx));
  if (lane < N_CLS) out[(size_t)s * N_CLS + lane] = acc[lane] - lse;
}

// ---------------------------------------------------------------------------
// Host orchestration
// ---------------------------------------------------------------------------
static inline size_t alignup(size_t x) { return (x + 255) & ~(size_t)255; }

extern "C" void kernel_launch(void* const* d_in, const int* in_sizes, int n_in,
                              void* d_out, int out_size, void* d_ws, size_t ws_size,
                              hipStream_t stream) {
  (void)in_sizes; (void)n_in; (void)out_size; (void)ws_size;
  const int*   indices = (const int*)d_in[0];
  const float* emb     = (const float*)d_in[1];
  const float* WQ      = (const float*)d_in[2];
  const float* WK      = (const float*)d_in[3];
  const float* WV      = (const float*)d_in[4];
  const float* WO_w    = (const float*)d_in[5];
  const float* WO_b    = (const float*)d_in[6];
  const float* FF1_w   = (const float*)d_in[7];
  const float* FF1_b   = (const float*)d_in[8];
  const float* FF2_w   = (const float*)d_in[9];
  const float* FF2_b   = (const float*)d_in[10];
  const float* Wc_w    = (const float*)d_in[11];
  const float* Wc_b    = (const float*)d_in[12];
  float* out = (float*)d_out;

  // workspace carve-up
  char* w = (char*)d_ws;
  auto alloc = [&](size_t bytes) { char* p = w; w += alignup(bytes); return (void*)p; };
  const size_t SD = (size_t)S_LEN * D_MODEL;
  bf16*  Wq_b  = (bf16*)alloc((size_t)D_MODEL * D_MODEL * 2);
  bf16*  Wk_b  = (bf16*)alloc((size_t)D_MODEL * D_MODEL * 2);
  bf16*  Wv_b  = (bf16*)alloc((size_t)D_MODEL * D_MODEL * 2);
  bf16*  Wo_b  = (bf16*)alloc((size_t)D_MODEL * D_MODEL * 2);
  bf16*  F1_b  = (bf16*)alloc((size_t)D_MODEL * D_FF * 2);
  bf16*  F2_b  = (bf16*)alloc((size_t)D_FF * D_MODEL * 2);
  float* Xf    = (float*)alloc(SD * 4);
  bf16*  Xh    = (bf16*)alloc(SD * 2);
  bf16*  Qh    = (bf16*)alloc(SD * 2);
  bf16*  KTt   = (bf16*)alloc(SD * 2);   // K transposed: [H*64, S]
  bf16*  Vh    = (bf16*)alloc(SD * 2);
  bf16*  Zc    = (bf16*)alloc(SD * 2);
  float* Att   = (float*)alloc(SD * 4);
  bf16*  Atth  = (bf16*)alloc(SD * 2);
  bf16*  Hh    = (bf16*)alloc((size_t)S_LEN * D_FF * 2);

  // 1) weight conversion
  {
    int nq = N_HEAD * D_MODEL * D_HEAD;  // 1M
    dim3 b(256), gq((nq + 255) / 256);
    cvt_qkv_kernel<<<gq, b, 0, stream>>>(WQ, Wq_b, N_HEAD, D_MODEL, D_HEAD);
    cvt_qkv_kernel<<<gq, b, 0, stream>>>(WK, Wk_b, N_HEAD, D_MODEL, D_HEAD);
    cvt_qkv_kernel<<<gq, b, 0, stream>>>(WV, Wv_b, N_HEAD, D_MODEL, D_HEAD);
    int no = D_MODEL * D_MODEL;
    cvt_plain_kernel<<<dim3((no + 255) / 256), b, 0, stream>>>(WO_w, Wo_b, no);
    int nf = D_MODEL * D_FF;
    cvt_plain_kernel<<<dim3((nf + 255) / 256), b, 0, stream>>>(FF1_w, F1_b, nf);
    cvt_plain_kernel<<<dim3((nf + 255) / 256), b, 0, stream>>>(FF2_w, F2_b, nf);
  }

  // 2) embedding gather
  embed_kernel<<<dim3(S_LEN), dim3(256), 0, stream>>>(indices, emb, Xf, Xh);

  // 3) two tied transformer layers
  const dim3 wave(32);
  const dim3 gFull(S_LEN / 32, D_MODEL / 64);
  const dim3 gFF1(S_LEN / 32, D_FF / 64);
  for (int layer = 0; layer < 2; ++layer) {
    gemm_bf16_kernel<false, false><<<gFull, wave, 0, stream>>>(
        Xh, Wq_b, nullptr, nullptr, nullptr, Qh, nullptr, S_LEN, D_MODEL, D_MODEL);
    gemm_bf16_kernel<false, false><<<gFull, wave, 0, stream>>>(
        Xh, Wk_b, nullptr, nullptr, nullptr, nullptr, KTt, S_LEN, D_MODEL, D_MODEL);
    gemm_bf16_kernel<false, false><<<gFull, wave, 0, stream>>>(
        Xh, Wv_b, nullptr, nullptr, nullptr, Vh, nullptr, S_LEN, D_MODEL, D_MODEL);

    flash_attn_kernel<<<dim3(S_LEN / 16, N_HEAD), wave, 0, stream>>>(Qh, KTt, Vh, Zc);

    // att = x + Zc @ WO + b   (f32 + bf16 mirrors)
    gemm_bf16_kernel<true, false><<<gFull, wave, 0, stream>>>(
        Zc, Wo_b, WO_b, Xf, Att, Atth, nullptr, S_LEN, D_MODEL, D_MODEL);

    // h = relu(att @ FF1 + b)
    gemm_bf16_kernel<true, true><<<gFF1, wave, 0, stream>>>(
        Atth, F1_b, FF1_b, nullptr, nullptr, Hh, nullptr, S_LEN, D_FF, D_MODEL);

    // x' = att + h @ FF2 + b
    gemm_bf16_kernel<true, false><<<gFull, wave, 0, stream>>>(
        Hh, F2_b, FF2_b, Att, Xf, Xh, nullptr, S_LEN, D_MODEL, D_FF);
  }

  // 4) classifier + log_softmax
  classifier_kernel<<<dim3(S_LEN), wave, 0, stream>>>(Xf, Wc_w, Wc_b, out);
}
